// MultiHeadAttention_84482006712419
// MI455X (gfx1250) — matrix operand order repaired and checked by reading, not compiled, not run on hardware
//
#include <hip/hip_runtime.h>
#include <hip/hip_bf16.h>

// ---------------- problem constants ----------------
constexpr int Bc = 2, Lc = 2048, Dc = 2048, Hc = 16, DHc = 128;
constexpr int Mrows = Bc * Lc;            // 4096
constexpr int QKVN = Bc * Lc * Dc;        // 8388608 elems
constexpr int WN   = Dc * Dc;             // 4194304 elems

typedef __attribute__((ext_vector_type(16))) __bf16 v16bf;
typedef __attribute__((ext_vector_type(8)))  __bf16 v8bf;
typedef __attribute__((ext_vector_type(8)))  float  v8f;

static __device__ __forceinline__ __bf16 f2bf(float f) {
  unsigned u = __builtin_bit_cast(unsigned, f);
  unsigned r = (u + 0x7FFFu + ((u >> 16) & 1u)) >> 16;
  unsigned short s = (unsigned short)r;
  return __builtin_bit_cast(__bf16, s);
}

static __device__ __forceinline__ v8f wmma_bf16(v16bf a, v16bf b, v8f c) {
  return __builtin_amdgcn_wmma_f32_16x16x32_bf16(false, a, false, b, (short)0, c, false, false);
}

static __device__ __forceinline__ v16bf cat8(v8bf lo, v8bf hi) {
  return __builtin_shufflevector(lo, hi, 0,1,2,3,4,5,6,7,8,9,10,11,12,13,14,15);
}

// ---------------- f32 -> bf16 conversion ----------------
__global__ void cvt_f32_bf16(const float* __restrict__ in, __bf16* __restrict__ out, int n) {
  int i = blockIdx.x * blockDim.x + threadIdx.x;
  if (i < n) out[i] = f2bf(in[i]);
}

// ---------------- GEMM:  C[m,n] = sum_k A[m,k] * W[n,k]  (x @ W^T) ----------------
// MODE 0: store bf16 row-major [Mrows, Dc]
// MODE 1: store bf16 transposed-per-head layout Vt[b][h][d][l]  (for P*V B-fragments)
// MODE 2: store f32 row-major [Mrows, Dc]  (final output)
template <int MODE>
__global__ __launch_bounds__(256)
void gemm_xWT(const __bf16* __restrict__ A, const __bf16* __restrict__ W,
              void* __restrict__ outp) {
  constexpr int K = Dc, N = Dc;
  const int lane = threadIdx.x & 31;
  const int wave = threadIdx.x >> 5;
  const int wm = wave & 3, wn = wave >> 2;           // 4 x 2 wave grid
  const int m0 = blockIdx.y * 128 + wm * 32;         // wave tile: 32 x 64
  const int n0 = blockIdx.x * 128 + wn * 64;
  const int half = lane >> 4, ln = lane & 15;

  v8f acc[2][4] = {};

  const __bf16* arow[2];
  const __bf16* wrow[4];
#pragma unroll
  for (int i = 0; i < 2; ++i) arow[i] = A + (size_t)(m0 + i * 16 + ln) * K + half * 16;
#pragma unroll
  for (int j = 0; j < 4; ++j) wrow[j] = W + (size_t)(n0 + j * 16 + ln) * K + half * 16;

  for (int k = 0; k < K; k += 32) {
    v16bf a[2], b[4];
#pragma unroll
    for (int i = 0; i < 2; ++i) {
      v8bf lo = *(const v8bf*)(arow[i] + k);
      v8bf hi = *(const v8bf*)(arow[i] + k + 8);
      a[i] = cat8(lo, hi);
    }
#pragma unroll
    for (int j = 0; j < 4; ++j) b[j] = *(const v16bf*)(wrow[j] + k);
#pragma unroll
    for (int i = 0; i < 2; ++i)
#pragma unroll
      for (int j = 0; j < 4; ++j) acc[i][j] = wmma_bf16(a[i], b[j], acc[i][j]);
  }

#pragma unroll
  for (int i = 0; i < 2; ++i)
#pragma unroll
    for (int j = 0; j < 4; ++j)
#pragma unroll
      for (int r = 0; r < 8; ++r) {
        const int row = m0 + i * 16 + half * 8 + r;
        const int col = n0 + j * 16 + ln;
        const float v = acc[i][j][r];
        if constexpr (MODE == 0) {
          ((__bf16*)outp)[(size_t)row * N + col] = f2bf(v);
        } else if constexpr (MODE == 1) {
          const int bb = row >> 11, l = row & (Lc - 1);
          const int h = col >> 7, d = col & (DHc - 1);
          ((__bf16*)outp)[(((size_t)(bb * Hc + h)) * DHc + d) * Lc + l] = f2bf(v);
        } else {
          ((float*)outp)[(size_t)row * N + col] = v;
        }
      }
}

// ---------------- flash attention (causal), per-wave 16 query rows ----------------
__global__ __launch_bounds__(128)
void flash_attn(const __bf16* __restrict__ Q, const __bf16* __restrict__ Kp,
                const __bf16* __restrict__ Vt, __bf16* __restrict__ Ctx) {
  __shared__ __bf16 Pbuf[4 * 16 * 32];                 // 1 KB per wave
  const int lane = threadIdx.x & 31, wave = threadIdx.x >> 5;
  const int half = lane >> 4, ln = lane & 15;
  constexpr int NQB = Lc / 64;                         // 32 query blocks per (b,h)
  const int qblk = blockIdx.x % NQB;
  const int bh = blockIdx.x / NQB;
  const int b = bh / Hc, h = bh % Hc;
  const int q0 = (qblk * 4 + wave) * 16;

  const __bf16* qbase = Q  + (size_t)b * Lc * Dc + h * DHc;
  const __bf16* kbase = Kp + (size_t)b * Lc * Dc + h * DHc;
  const __bf16* vbase = Vt + (size_t)(b * Hc + h) * DHc * Lc;
  __bf16* pw = Pbuf + wave * 512;

  // preload the 4 Q A-fragments (DH = 128 = 4 * 32)
  v16bf qa[4];
#pragma unroll
  for (int dk = 0; dk < 4; ++dk) {
    const __bf16* ap = qbase + (size_t)(q0 + ln) * Dc + dk * 32 + half * 16;
    qa[dk] = cat8(*(const v8bf*)ap, *(const v8bf*)(ap + 8));
  }

  v8f ctx[8] = {};
  float rm[8], rs[8];
#pragma unroll
  for (int r = 0; r < 8; ++r) { rm[r] = -__builtin_inff(); rs[r] = 0.f; }

  const float scale = 0.08838834764831845f;            // 1/sqrt(128)
  const int kmax = q0 + 16;                            // exclusive key bound

  for (int kb = 0; kb < kmax; kb += 32) {
    // ---- scores for 32 keys: two 16x16 C fragments ----
    v8f s0 = {}, s1 = {};
    const __bf16* kc = kbase + (size_t)(kb + ln) * Dc + half * 16;
#pragma unroll
    for (int dk = 0; dk < 4; ++dk) {
      v16bf b0 = *(const v16bf*)(kc + dk * 32);
      s0 = wmma_bf16(qa[dk], b0, s0);
      v16bf b1 = *(const v16bf*)(kc + (size_t)16 * Dc + dk * 32);
      s1 = wmma_bf16(qa[dk], b1, s1);
    }

    // ---- scale + causal mask + online softmax ----
    const bool needmask = (kb + 31 > q0);
    const int key0 = kb + ln, key1 = kb + 16 + ln;
    float p0[8], p1[8], t[8];
#pragma unroll
    for (int r = 0; r < 8; ++r) {
      const int row = q0 + half * 8 + r;
      float a0 = s0[r] * scale, a1 = s1[r] * scale;
      if (needmask) {
        if (key0 > row) a0 = -__builtin_inff();
        if (key1 > row) a1 = -__builtin_inff();
      }
      p0[r] = a0; p1[r] = a1;
      t[r] = fmaxf(a0, a1);
    }
#pragma unroll
    for (int off = 1; off < 16; off <<= 1)
#pragma unroll
      for (int r = 0; r < 8; ++r) t[r] = fmaxf(t[r], __shfl_xor(t[r], off, 32));

    float fr[8];
#pragma unroll
    for (int r = 0; r < 8; ++r) {
      const float mn = fmaxf(rm[r], t[r]);
      fr[r] = __expf(rm[r] - mn);
      rm[r] = mn;
      p0[r] = __expf(p0[r] - mn);
      p1[r] = __expf(p1[r] - mn);
    }
    float ls[8];
#pragma unroll
    for (int r = 0; r < 8; ++r) ls[r] = p0[r] + p1[r];
#pragma unroll
    for (int off = 1; off < 16; off <<= 1)
#pragma unroll
      for (int r = 0; r < 8; ++r) ls[r] += __shfl_xor(ls[r], off, 32);
#pragma unroll
    for (int r = 0; r < 8; ++r) rs[r] = rs[r] * fr[r] + ls[r];

    // rescale running context accumulators
#pragma unroll
    for (int dt = 0; dt < 8; ++dt)
#pragma unroll
      for (int r = 0; r < 8; ++r) ctx[dt][r] *= fr[r];

    // ---- C-layout -> A-layout relayout of P through per-wave LDS ----
#pragma unroll
    for (int r = 0; r < 8; ++r) {
      pw[(half * 8 + r) * 32 + ln]      = f2bf(p0[r]);
      pw[(half * 8 + r) * 32 + 16 + ln] = f2bf(p1[r]);
    }
    const __bf16* pp = pw + ln * 32 + half * 16;       // A-frag: k chunks {h*16, h*16+8}
    v16bf pa = cat8(*(const v8bf*)pp, *(const v8bf*)(pp + 8));

    // ---- P (16x32) x V (32x16 per d-tile), V pre-transposed so loads are contiguous ----
    const __bf16* vc = vbase + (size_t)ln * Lc + kb + half * 16;
#pragma unroll
    for (int dt = 0; dt < 8; ++dt) {
      v16bf vf = *(const v16bf*)(vc + (size_t)dt * 16 * Lc);
      ctx[dt] = wmma_bf16(pa, vf, ctx[dt]);
    }
  }

  // ---- normalize + store context (merged heads, row-major [B,L,D]) ----
  float inv[8];
#pragma unroll
  for (int r = 0; r < 8; ++r) inv[r] = 1.0f / rs[r];
#pragma unroll
  for (int dt = 0; dt < 8; ++dt)
#pragma unroll
    for (int r = 0; r < 8; ++r) {
      const int row = q0 + half * 8 + r;
      Ctx[((size_t)b * Lc + row) * Dc + h * DHc + dt * 16 + ln] = f2bf(ctx[dt][r] * inv[r]);
    }
}

// ---------------- host side ----------------
extern "C" void kernel_launch(void* const* d_in, const int* in_sizes, int n_in,
                              void* d_out, int out_size, void* d_ws, size_t ws_size,
                              hipStream_t stream) {
  const float* q  = (const float*)d_in[0];
  const float* k  = (const float*)d_in[1];
  const float* v  = (const float*)d_in[2];
  // d_in[3] = attn_mask (causal; handled analytically)
  const float* wq = (const float*)d_in[4];
  const float* wk = (const float*)d_in[5];
  const float* wv = (const float*)d_in[6];
  const float* wo = (const float*)d_in[7];

  char* ws = (char*)d_ws;
  size_t off = 0;
  auto alloc = [&](size_t bytes) { char* p = ws + off; off += (bytes + 255) & ~size_t(255); return p; };
  __bf16* qb  = (__bf16*)alloc((size_t)QKVN * 2);
  __bf16* kb  = (__bf16*)alloc((size_t)QKVN * 2);
  __bf16* vb  = (__bf16*)alloc((size_t)QKVN * 2);
  __bf16* wqb = (__bf16*)alloc((size_t)WN * 2);
  __bf16* wkb = (__bf16*)alloc((size_t)WN * 2);
  __bf16* wvb = (__bf16*)alloc((size_t)WN * 2);
  __bf16* wob = (__bf16*)alloc((size_t)WN * 2);
  __bf16* qp  = (__bf16*)alloc((size_t)QKVN * 2);   // projected Q, [B,L,D]
  __bf16* kp  = (__bf16*)alloc((size_t)QKVN * 2);   // projected K, [B,L,D]
  __bf16* vt  = (__bf16*)alloc((size_t)QKVN * 2);   // projected V, [B,H,DH,L]
  __bf16* ctx = (__bf16*)alloc((size_t)QKVN * 2);   // attention output, [B,L,D]

  const int T = 256;
  cvt_f32_bf16<<<(QKVN + T - 1) / T, T, 0, stream>>>(q, qb, QKVN);
  cvt_f32_bf16<<<(QKVN + T - 1) / T, T, 0, stream>>>(k, kb, QKVN);
  cvt_f32_bf16<<<(QKVN + T - 1) / T, T, 0, stream>>>(v, vb, QKVN);
  cvt_f32_bf16<<<(WN   + T - 1) / T, T, 0, stream>>>(wq, wqb, WN);
  cvt_f32_bf16<<<(WN   + T - 1) / T, T, 0, stream>>>(wk, wkb, WN);
  cvt_f32_bf16<<<(WN   + T - 1) / T, T, 0, stream>>>(wv, wvb, WN);
  cvt_f32_bf16<<<(WN   + T - 1) / T, T, 0, stream>>>(wo, wob, WN);

  dim3 ggrid(Dc / 128, Mrows / 128);                  // 16 x 32
  gemm_xWT<0><<<ggrid, 256, 0, stream>>>(qb, wqb, (void*)qp);
  gemm_xWT<0><<<ggrid, 256, 0, stream>>>(kb, wkb, (void*)kp);
  gemm_xWT<1><<<ggrid, 256, 0, stream>>>(vb, wvb, (void*)vt);

  flash_attn<<<Bc * Hc * (Lc / 64), 128, 0, stream>>>(qp, kp, vt, ctx);

  gemm_xWT<2><<<ggrid, 256, 0, stream>>>(ctx, wob, d_out);
}
